// UberCRSN_24902220382705
// MI455X (gfx1250) — matrix operand order repaired and checked by reading, compiled, not run on hardware
//
#include <hip/hip_runtime.h>
#include <hip/hip_bf16.h>
#include <stdint.h>

typedef __attribute__((ext_vector_type(16))) _Float16 v16h;
typedef __attribute__((ext_vector_type(8)))  _Float16 v8h;
typedef __attribute__((ext_vector_type(8)))  float    v8f;

// =====================================================================
// WMMA GEMM:  C[m][n] = alpha * sum_k A[m][k]*B[n][k]  (+ beta*C + bias)
// A: f16 row-major (lda), B: f16 row-major (ldb, K contiguous), C: f32 (ldc)
// Requires M%128==0, N%128==0, K%32==0 (true for every matmul here).
// Block: 256 threads = 8 waves stacked along M; block tile 128Mx128N;
// wave tile 16Mx128N (8 f32 accumulators = 64 VGPRs).
// Inner loop is pure loads + v_wmma (no VALU conversions).
// =====================================================================
__global__ __launch_bounds__(256) void gemm_wmma_f16(
    const _Float16* __restrict__ A, const _Float16* __restrict__ B, float* __restrict__ C,
    int K, int lda, int ldb, int ldc,
    long long sA, long long sB, long long sC,
    const float* __restrict__ bias, float alpha, float beta)
{
  const int wid  = threadIdx.x >> 5;
  const int lane = threadIdx.x & 31;
  const int half = lane >> 4;          // K-half selector (contiguous k = 16*half + s, same for A and B)
  const int r    = lane & 15;          // row (A) / column (B,C) within tile
  const int m0 = blockIdx.x * 128 + wid * 16;
  const int n0 = blockIdx.y * 128;
  const long long zb = blockIdx.z;
  A += zb * sA; B += zb * sB; C += zb * sC;

  v8f acc[8] = {};
  const _Float16* arow = A + (long long)(m0 + r) * lda + half * 16;
  const _Float16* brow[8];
#pragma unroll
  for (int t = 0; t < 8; ++t)
    brow[t] = B + (long long)(n0 + 16 * t + r) * ldb + half * 16;

  for (int k0 = 0; k0 < K; k0 += 32) {
    // A fragment: 16 f16/lane; K = half*16 + 0..15
    v8h lo = *(const v8h*)(arow + k0);
    v8h hi = *(const v8h*)(arow + k0 + 8);
    __builtin_prefetch(arow + k0 + 64, 0, 1);
    v16h a = __builtin_shufflevector(lo, hi, 0,1,2,3,4,5,6,7,8,9,10,11,12,13,14,15);
    // 8 B fragments: one 32B contiguous load each; K = half*16 + 0..15
#pragma unroll
    for (int t = 0; t < 8; ++t) {
      v16h b = *(const v16h*)(brow[t] + k0);
      acc[t] = __builtin_amdgcn_wmma_f32_16x16x32_f16(
                 false, a, false, b, (short)0, acc[t], false, false);
    }
  }

#pragma unroll
  for (int t = 0; t < 8; ++t) {
    const int n = n0 + 16 * t + r;
    const float bv = bias ? bias[n] : 0.0f;
#pragma unroll
    for (int i = 0; i < 8; ++i) {
      long long off = (long long)(m0 + half * 8 + i) * ldc + n;
      float v = fmaf(alpha, acc[t][i], bv);
      if (beta != 0.0f) v = fmaf(beta, C[off], v);
      C[off] = v;
    }
  }
}

// ============================ utility kernels ============================
__global__ void fill_f32(float* p, float v, long long n) {
  long long i = (long long)blockIdx.x * blockDim.x + threadIdx.x;
  if (i < n) p[i] = v;
}

__global__ void cvt_f16(const float* __restrict__ s, _Float16* __restrict__ d, long long n) {
  long long i = (long long)blockIdx.x * blockDim.x + threadIdx.x;
  if (i < n) d[i] = (_Float16)s[i];
}

__global__ void embed_gather(const int* __restrict__ tok,
                             const float* __restrict__ er, const float* __restrict__ ei,
                             _Float16* __restrict__ zr, _Float16* __restrict__ zi, int N, int D) {
  long long i = (long long)blockIdx.x * blockDim.x + threadIdx.x;
  if (i >= (long long)N * D) return;
  long long n = i / D; int d = (int)(i % D);
  int t = tok[n];
  zr[i] = (_Float16)er[(long long)t * D + d];
  zi[i] = (_Float16)ei[(long long)t * D + d];
}

// magnitude-LayerNorm + phase restore + modReLU. Reads f32, writes f16 (GEMM A).
__global__ __launch_bounds__(256) void magln_modrelu(
    const float* __restrict__ pr, const float* __restrict__ pi,
    _Float16* __restrict__ pr16, _Float16* __restrict__ pi16,
    const float* __restrict__ lns, const float* __restrict__ lnb,
    const float* __restrict__ mb, int D)
{
  long long n = blockIdx.x;
  int d = threadIdx.x;
  long long off = n * D + d;
  float x = pr[off], y = pi[off];
  float h = sqrtf(x * x + y * y);
  float mag = h + 1e-6f;
  __shared__ float red[256];
  red[d] = mag; __syncthreads();
  for (int s = 128; s > 0; s >>= 1) { if (d < s) red[d] += red[d + s]; __syncthreads(); }
  float mean = red[0] / D; __syncthreads();
  float dm = mag - mean;
  red[d] = dm * dm; __syncthreads();
  for (int s = 128; s > 0; s >>= 1) { if (d < s) red[d] += red[d + s]; __syncthreads(); }
  float var = red[0] / (D - 1);
  float nmag = dm * rsqrtf(var + 1e-6f) * lns[d] + lnb[d];
  // cos/sin(atan2(y,x)) = x/h, y/h  (atan2(0,0)=0 -> cos=1,sin=0)
  float inv = (h > 0.f) ? (1.f / h) : 0.f;
  float cr = (h > 0.f) ? x * inv : 1.f;
  float ci = y * inv;
  float prn = nmag * cr, pin = nmag * ci;
  float nrm = sqrtf(prn * prn + pin * pin) + 1e-6f;
  float sc = fmaxf(nrm + mb[d], 0.f) / nrm;
  pr16[off] = (_Float16)(prn * sc);
  pi16[off] = (_Float16)(pin * sc);
}

// concat(r,i) -> f16 (GEMM operand)
__global__ void pack_concat(const float* __restrict__ r, const float* __restrict__ im,
                            _Float16* __restrict__ o16, int N, int D) {
  long long i = (long long)blockIdx.x * blockDim.x + threadIdx.x;
  if (i >= (long long)N * 2 * D) return;
  long long n = i / (2 * D); int d = (int)(i % (2 * D));
  float v = (d < D) ? r[n * D + d] : im[n * D + (d - D)];
  o16[i] = (_Float16)v;
}

// per-batch [S][D] f32 -> [D][S] f16 (grid.z = batch)
__global__ __launch_bounds__(1024) void transpose_f16(
    const float* __restrict__ src, _Float16* __restrict__ dst, int S, int D) {
  __shared__ float tile[32][33];
  long long b = blockIdx.z;
  const float* sp = src + b * (long long)S * D;
  _Float16* dp = dst + b * (long long)S * D;
  int s0 = blockIdx.x * 32, d0 = blockIdx.y * 32;
  int tx = threadIdx.x & 31, ty = threadIdx.x >> 5;
  tile[ty][tx] = sp[(long long)(s0 + ty) * D + d0 + tx];
  __syncthreads();
  dp[(long long)(d0 + ty) * S + s0 + tx] = (_Float16)tile[tx][ty];
}

// softmax over rows of length 512 (scale pre-exp); writes f16 for PV GEMM A
__global__ __launch_bounds__(256) void softmax512(const float* __restrict__ attn,
                                                  _Float16* __restrict__ attn16, float scale) {
  long long row = blockIdx.x;
  const float* p = attn + row * 512;
  _Float16* o = attn16 + row * 512;
  int t = threadIdx.x;
  float a = p[t] * scale, b = p[t + 256] * scale;
  __shared__ float red[256];
  red[t] = fmaxf(a, b); __syncthreads();
  for (int s = 128; s > 0; s >>= 1) { if (t < s) red[t] = fmaxf(red[t], red[t + s]); __syncthreads(); }
  float mx = red[0]; __syncthreads();
  float ea = __expf(a - mx), eb = __expf(b - mx);
  red[t] = ea + eb; __syncthreads();
  for (int s = 128; s > 0; s >>= 1) { if (t < s) red[t] += red[t + s]; __syncthreads(); }
  float inv = 1.0f / red[0];
  o[t] = (_Float16)(ea * inv); o[t + 256] = (_Float16)(eb * inv);
}

// halt + ctrl heads: one wave per token (wave32 shuffle reduction)
__global__ __launch_bounds__(256) void halt_ctrl(
    const float* __restrict__ ar, const float* __restrict__ ai,
    const float* __restrict__ Wh, const float* __restrict__ bh,
    const float* __restrict__ Ws, const float* __restrict__ bs,
    float* __restrict__ halt, float* __restrict__ ctrl, int N, int D)
{
  int wid = threadIdx.x >> 5, lane = threadIdx.x & 31;
  int n = blockIdx.x * 8 + wid;
  if (n >= N) return;
  float h = 0.f, s0 = 0.f, s1 = 0.f, s2 = 0.f;
#pragma unroll
  for (int j = 0; j < 16; ++j) {
    int e = lane + j * 32;  // 0..511
    float z = (e < D) ? ar[(long long)n * D + e] : ai[(long long)n * D + (e - D)];
    h  += z * Wh[e];
    s0 += z * Ws[e];
    s1 += z * Ws[512 + e];
    s2 += z * Ws[1024 + e];
  }
  for (int m = 16; m >= 1; m >>= 1) {
    h  += __shfl_xor(h, m);  s0 += __shfl_xor(s0, m);
    s1 += __shfl_xor(s1, m); s2 += __shfl_xor(s2, m);
  }
  if (lane == 0) {
    float hv = 1.f / (1.f + __expf(-(h + bh[0])));
    float a0 = s0 + bs[0], a1 = s1 + bs[1], a2 = s2 + bs[2];
    float mx = fmaxf(a0, fmaxf(a1, a2));
    float e0 = __expf(a0 - mx), e1 = __expf(a1 - mx), e2 = __expf(a2 - mx);
    float inv = 1.f / (e0 + e1 + e2);
    halt[n] = hv;
    ctrl[n * 3 + 0] = e0 * inv; ctrl[n * 3 + 1] = e1 * inv; ctrl[n * 3 + 2] = e2 * inv;
  }
}

__global__ void init_ptr(float* ptr, int N) {
  int n = blockIdx.x * blockDim.x + threadIdx.x;
  if (n >= N) return;
  for (int k = 0; k < 8; ++k) ptr[n * 8 + k] = (k == 0) ? 1.f : 0.f;
}

// stack pointer roll/blend/normalize; wmask from OLD ptr
__global__ void stack_ptr_update(const float* __restrict__ ctrl, const float* __restrict__ ptr,
                                 float* __restrict__ nptr, float* __restrict__ wmask, int N) {
  int n = blockIdx.x * blockDim.x + threadIdx.x;
  if (n >= N) return;
  float push = ctrl[n * 3], pop = ctrl[n * 3 + 1], noop = ctrl[n * 3 + 2];
  float p[8];
#pragma unroll
  for (int k = 0; k < 8; ++k) p[k] = ptr[n * 8 + k];
  float np[8], s = 0.f;
#pragma unroll
  for (int k = 0; k < 8; ++k) {
    float up = p[(k + 7) & 7], dn = p[(k + 1) & 7];
    float v = push * up + pop * dn + noop * p[k];
    np[k] = v; s += v;
    wmask[n * 8 + k] = push * up;
  }
  float inv = 1.f / (s + 1e-6f);
#pragma unroll
  for (int k = 0; k < 8; ++k) nptr[n * 8 + k] = np[k] * inv;
}

// memory blend + weighted read; one block (512 thr) per token.
// Writes f32 read (for accumulation) + f16 mirror (next clinear A operand).
__global__ __launch_bounds__(512) void stack_mem_read(
    const float* __restrict__ wmask, const float* __restrict__ nptr,
    const float* __restrict__ ar, const float* __restrict__ ai,
    float* __restrict__ memory,
    float* __restrict__ rr, float* __restrict__ ri,
    _Float16* __restrict__ rr16, _Float16* __restrict__ ri16, int D)
{
  long long n = blockIdx.x;
  int d = threadIdx.x;  // 0..511
  float zf = (d < D) ? ar[n * D + d] : ai[n * D + (d - D)];
  float acc = 0.f;
  long long base = n * 8 * 512 + d;
#pragma unroll
  for (int k = 0; k < 8; ++k) {
    float w = wmask[n * 8 + k];
    float m = memory[base + k * 512];
    m = fmaf(w, zf, m * (1.f - w));
    memory[base + k * 512] = m;
    acc = fmaf(nptr[n * 8 + k], m, acc);
  }
  if (d < D) { rr[n * D + d] = acc; rr16[n * D + d] = (_Float16)acc; }
  else       { ri[n * D + (d - D)] = acc; ri16[n * D + (d - D)] = (_Float16)acc; }
}

__global__ void accum_step(float* __restrict__ accr, float* __restrict__ acci,
                           const float* __restrict__ rr, const float* __restrict__ ri,
                           const float* __restrict__ halt, const float* __restrict__ nh,
                           int last, int N, int D) {
  long long i = (long long)blockIdx.x * blockDim.x + threadIdx.x;
  if (i >= (long long)N * D) return;
  long long n = i / D;
  float w = nh[n] * (last ? 1.f : halt[n]);
  accr[i] = fmaf(w, rr[i], accr[i]);
  acci[i] = fmaf(w, ri[i], acci[i]);
}

__global__ void update_nh(float* nh, const float* halt, int N) {
  int n = blockIdx.x * blockDim.x + threadIdx.x;
  if (n < N) nh[n] *= (1.f - halt[n]);
}

__global__ __launch_bounds__(256) void cb_norms(const float* __restrict__ cb, float* __restrict__ cbn, int K) {
  long long j = blockIdx.x;
  int t = threadIdx.x;
  float s = 0.f;
  for (int k = t; k < K; k += 256) { float v = cb[j * K + k]; s += v * v; }
  __shared__ float red[256];
  red[t] = s; __syncthreads();
  for (int st = 128; st > 0; st >>= 1) { if (t < st) red[t] += red[t + st]; __syncthreads(); }
  if (t == 0) cbn[j] = 0.5f * red[0];
}

// argmin_j (cbn[j] - score[n][j])  == argmin distance
__global__ __launch_bounds__(256) void argmin_rows(const float* __restrict__ scores,
                                                   const float* __restrict__ cbn,
                                                   int* __restrict__ idx, int NS) {
  long long n = blockIdx.x;
  int t = threadIdx.x;
  float best = 3.4e38f; int bj = 0x7fffffff;
  for (int j = t; j < NS; j += 256) {
    float v = cbn[j] - scores[n * NS + j];
    if (v < best || (v == best && j < bj)) { best = v; bj = j; }
  }
  __shared__ float rb[256]; __shared__ int rj[256];
  rb[t] = best; rj[t] = bj; __syncthreads();
  for (int s = 128; s > 0; s >>= 1) {
    if (t < s && (rb[t + s] < rb[t] || (rb[t + s] == rb[t] && rj[t + s] < rj[t]))) {
      rb[t] = rb[t + s]; rj[t] = rj[t + s];
    }
    __syncthreads();
  }
  if (t == 0) idx[n] = rj[0];
}

__global__ __launch_bounds__(256) void gather_cb(const int* __restrict__ idx,
                                                 const float* __restrict__ cb,
                                                 _Float16* __restrict__ zq16, int K) {
  long long n = blockIdx.x;
  int j = idx[n];
  for (int k = threadIdx.x; k < K; k += 256)
    zq16[n * K + k] = (_Float16)cb[(long long)j * K + k];
}

// ============================== host side ==============================
extern "C" void kernel_launch(void* const* d_in, const int* in_sizes, int n_in,
                              void* d_out, int out_size, void* d_ws, size_t ws_size,
                              hipStream_t stream) {
  (void)in_sizes; (void)n_in; (void)out_size; (void)ws_size;
  const int Bb = 16, S = 512, D = 256, NS = 1024;
  const int N = Bb * S;          // 8192
  const int D2 = 2 * D;          // 512
  const int DEPTHN = 4;

  const int*   tokens  = (const int*)  d_in[0];
  const float* emb_r   = (const float*)d_in[1];
  const float* emb_i   = (const float*)d_in[2];
  const float* Wl_r    = (const float*)d_in[3];
  const float* Wl_i    = (const float*)d_in[4];
  const float* ln_s    = (const float*)d_in[5];
  const float* ln_b    = (const float*)d_in[6];
  const float* mod_b   = (const float*)d_in[7];
  const float* Wq_r    = (const float*)d_in[8];
  const float* Wq_i    = (const float*)d_in[9];
  const float* Wk_r    = (const float*)d_in[10];
  const float* Wk_i    = (const float*)d_in[11];
  const float* Wv_r    = (const float*)d_in[12];
  const float* Wv_i    = (const float*)d_in[13];
  const float* Wh      = (const float*)d_in[14];
  const float* bh      = (const float*)d_in[15];
  const float* Ws      = (const float*)d_in[16];
  const float* bs      = (const float*)d_in[17];
  const float* codebook= (const float*)d_in[18];
  const float* Wout    = (const float*)d_in[19];
  const float* bout    = (const float*)d_in[20];
  float* out = (float*)d_out;

  // ---- bump allocator over workspace (256B aligned) ----
  char* wp = (char*)d_ws;
  auto alloc = [&](size_t bytes) -> void* {
    uintptr_t p = ((uintptr_t)wp + 255) & ~(uintptr_t)255;
    wp = (char*)(p + bytes);
    return (void*)p;
  };
  const size_t ND  = (size_t)N * D;   // 2M elems
  const size_t ND2 = (size_t)N * D2;  // 4M elems

  // f16 activations (GEMM A/B operands)
  _Float16* zr16 = (_Float16*)alloc(ND * 2);
  _Float16* zi16 = (_Float16*)alloc(ND * 2);
  _Float16* rr16 = (_Float16*)alloc(ND * 2);
  _Float16* ri16 = (_Float16*)alloc(ND * 2);
  _Float16* pr16 = (_Float16*)alloc(ND * 2);
  _Float16* pi16 = (_Float16*)alloc(ND * 2);
  _Float16* qf16 = (_Float16*)alloc(ND2 * 2);
  _Float16* kf16 = (_Float16*)alloc(ND2 * 2);
  _Float16* vrT16 = (_Float16*)alloc(ND * 2);
  _Float16* viT16 = (_Float16*)alloc(ND * 2);
  _Float16* attn16 = (_Float16*)alloc((size_t)Bb * S * S * 2);
  _Float16* zqin16 = (_Float16*)alloc(ND2 * 2);
  _Float16* zq16   = (_Float16*)alloc(ND2 * 2);
  // f32 intermediates
  float* pr   = (float*)alloc(ND * 4);   // clinear out / attention out (real)
  float* pi   = (float*)alloc(ND * 4);   // clinear out / attention out (imag)
  float* qr   = (float*)alloc(ND * 4);
  float* qi   = (float*)alloc(ND * 4);
  float* kr   = (float*)alloc(ND * 4);
  float* ki   = (float*)alloc(ND * 4);
  float* vr   = (float*)alloc(ND * 4);
  float* vi   = (float*)alloc(ND * 4);
  float* rr   = (float*)alloc(ND * 4);
  float* ri   = (float*)alloc(ND * 4);
  float* attn = (float*)alloc((size_t)Bb * S * S * 4);
  float* memv = (float*)alloc((size_t)N * 8 * D2 * 4);
  float* ptr  = (float*)alloc((size_t)N * 8 * 4);
  float* nptr = (float*)alloc((size_t)N * 8 * 4);
  float* wmask= (float*)alloc((size_t)N * 8 * 4);
  float* accr = (float*)alloc(ND * 4);
  float* acci = (float*)alloc(ND * 4);
  float* halt = (float*)alloc((size_t)N * 4);
  float* nh   = (float*)alloc((size_t)N * 4);
  float* ctrl = (float*)alloc((size_t)N * 3 * 4);
  float* scores = (float*)alloc((size_t)N * NS * 4);
  float* cbn  = (float*)alloc((size_t)NS * 4);
  int*   idx  = (int*)alloc((size_t)N * 4);
  // f16 weights
  _Float16* wWl_r = (_Float16*)alloc((size_t)D * D * 2);
  _Float16* wWl_i = (_Float16*)alloc((size_t)D * D * 2);
  _Float16* wWq_r = (_Float16*)alloc((size_t)D * D * 2);
  _Float16* wWq_i = (_Float16*)alloc((size_t)D * D * 2);
  _Float16* wWk_r = (_Float16*)alloc((size_t)D * D * 2);
  _Float16* wWk_i = (_Float16*)alloc((size_t)D * D * 2);
  _Float16* wWv_r = (_Float16*)alloc((size_t)D * D * 2);
  _Float16* wWv_i = (_Float16*)alloc((size_t)D * D * 2);
  _Float16* cb16  = (_Float16*)alloc((size_t)NS * D2 * 2);
  _Float16* wo16  = (_Float16*)alloc((size_t)256 * D2 * 2);

  auto blocks = [](long long total, int thr) { return (unsigned)((total + thr - 1) / thr); };
  auto gemm = [&](const _Float16* A, const _Float16* Bm, float* C, int M, int Nn, int K,
                  int lda, int ldb, int ldc, long long sA, long long sB, long long sC,
                  int batches, const float* bias, float alpha, float beta) {
    dim3 g(M / 128, Nn / 128, batches);
    gemm_wmma_f16<<<g, 256, 0, stream>>>(A, Bm, C, K, lda, ldb, ldc, sA, sB, sC,
                                         bias, alpha, beta);
  };

  // ---- weight conversion to f16 ----
  long long dd = (long long)D * D;
  cvt_f16<<<blocks(dd,256),256,0,stream>>>(Wl_r, wWl_r, dd);
  cvt_f16<<<blocks(dd,256),256,0,stream>>>(Wl_i, wWl_i, dd);
  cvt_f16<<<blocks(dd,256),256,0,stream>>>(Wq_r, wWq_r, dd);
  cvt_f16<<<blocks(dd,256),256,0,stream>>>(Wq_i, wWq_i, dd);
  cvt_f16<<<blocks(dd,256),256,0,stream>>>(Wk_r, wWk_r, dd);
  cvt_f16<<<blocks(dd,256),256,0,stream>>>(Wk_i, wWk_i, dd);
  cvt_f16<<<blocks(dd,256),256,0,stream>>>(Wv_r, wWv_r, dd);
  cvt_f16<<<blocks(dd,256),256,0,stream>>>(Wv_i, wWv_i, dd);
  cvt_f16<<<blocks((long long)NS*D2,256),256,0,stream>>>(codebook, cb16, (long long)NS*D2);
  cvt_f16<<<blocks((long long)256*D2,256),256,0,stream>>>(Wout, wo16, (long long)256*D2);
  cb_norms<<<NS,256,0,stream>>>(codebook, cbn, D2);

  // ---- state init ----
  embed_gather<<<blocks((long long)ND,256),256,0,stream>>>(tokens, emb_r, emb_i, zr16, zi16, N, D);
  fill_f32<<<blocks((long long)N*8*D2,256),256,0,stream>>>(memv, 0.f, (long long)N*8*D2);
  init_ptr<<<blocks(N,256),256,0,stream>>>(ptr, N);
  fill_f32<<<blocks((long long)ND,256),256,0,stream>>>(accr, 0.f, ND);
  fill_f32<<<blocks((long long)ND,256),256,0,stream>>>(acci, 0.f, ND);
  fill_f32<<<blocks(N,256),256,0,stream>>>(nh, 1.f, N);

  const float sc = 1.0f / sqrtf((float)D);

  for (int t = 0; t < DEPTHN; ++t) {
    // complex linear: p = z @ Wl^T (complex)
    gemm(zr16, wWl_r, pr, N, D, D, D, D, D, 0,0,0, 1, nullptr,  1.f, 0.f);
    gemm(zi16, wWl_i, pr, N, D, D, D, D, D, 0,0,0, 1, nullptr, -1.f, 1.f);
    gemm(zi16, wWl_r, pi, N, D, D, D, D, D, 0,0,0, 1, nullptr,  1.f, 0.f);
    gemm(zr16, wWl_i, pi, N, D, D, D, D, D, 0,0,0, 1, nullptr,  1.f, 1.f);
    // magnitude LN + modReLU -> f16 operands for Q/K/V
    magln_modrelu<<<N,256,0,stream>>>(pr, pi, pr16, pi16, ln_s, ln_b, mod_b, D);
    // Q, K, V complex linears
    gemm(pr16, wWq_r, qr, N, D, D, D, D, D, 0,0,0, 1, nullptr,  1.f, 0.f);
    gemm(pi16, wWq_i, qr, N, D, D, D, D, D, 0,0,0, 1, nullptr, -1.f, 1.f);
    gemm(pi16, wWq_r, qi, N, D, D, D, D, D, 0,0,0, 1, nullptr,  1.f, 0.f);
    gemm(pr16, wWq_i, qi, N, D, D, D, D, D, 0,0,0, 1, nullptr,  1.f, 1.f);
    gemm(pr16, wWk_r, kr, N, D, D, D, D, D, 0,0,0, 1, nullptr,  1.f, 0.f);
    gemm(pi16, wWk_i, kr, N, D, D, D, D, D, 0,0,0, 1, nullptr, -1.f, 1.f);
    gemm(pi16, wWk_r, ki, N, D, D, D, D, D, 0,0,0, 1, nullptr,  1.f, 0.f);
    gemm(pr16, wWk_i, ki, N, D, D, D, D, D, 0,0,0, 1, nullptr,  1.f, 1.f);
    gemm(pr16, wWv_r, vr, N, D, D, D, D, D, 0,0,0, 1, nullptr,  1.f, 0.f);
    gemm(pi16, wWv_i, vr, N, D, D, D, D, D, 0,0,0, 1, nullptr, -1.f, 1.f);
    gemm(pi16, wWv_r, vi, N, D, D, D, D, D, 0,0,0, 1, nullptr,  1.f, 0.f);
    gemm(pr16, wWv_i, vi, N, D, D, D, D, D, 0,0,0, 1, nullptr,  1.f, 1.f);
    // pack qf/kf (f16); transpose V to f16
    pack_concat<<<blocks((long long)ND2,256),256,0,stream>>>(qr, qi, qf16, N, D);
    pack_concat<<<blocks((long long)ND2,256),256,0,stream>>>(kr, ki, kf16, N, D);
    {
      dim3 tg(S/32, D/32, Bb);
      transpose_f16<<<tg,1024,0,stream>>>(vr, vrT16, S, D);
      transpose_f16<<<tg,1024,0,stream>>>(vi, viT16, S, D);
    }
    // attention: scores (batched), softmax -> f16 probs, PV (batched) -> pr, pi reused
    gemm(qf16, kf16, attn, S, S, D2, D2, D2, S,
         (long long)S*D2, (long long)S*D2, (long long)S*S, Bb, nullptr, 1.f, 0.f);
    softmax512<<<Bb*S,256,0,stream>>>(attn, attn16, sc);
    gemm(attn16, vrT16, pr, S, D, S, S, S, D,
         (long long)S*S, (long long)D*S, (long long)S*D, Bb, nullptr, 1.f, 0.f);
    gemm(attn16, viT16, pi, S, D, S, S, S, D,
         (long long)S*S, (long long)D*S, (long long)S*D, Bb, nullptr, 1.f, 0.f);
    // halt / ctrl heads
    halt_ctrl<<<N/8,256,0,stream>>>(pr, pi, Wh, bh, Ws, bs, halt, ctrl, N, D);
    // stack update + read (f32 read + f16 mirror for next depth's A operand)
    stack_ptr_update<<<blocks(N,256),256,0,stream>>>(ctrl, ptr, nptr, wmask, N);
    stack_mem_read<<<N,512,0,stream>>>(wmask, nptr, pr, pi, memv, rr, ri, rr16, ri16, D);
    { float* tmp = ptr; ptr = nptr; nptr = tmp; }
    // accumulate + halting state
    accum_step<<<blocks((long long)ND,256),256,0,stream>>>(accr, acci, rr, ri, halt, nh,
                                                           (t == DEPTHN-1) ? 1 : 0, N, D);
    update_nh<<<blocks(N,256),256,0,stream>>>(nh, halt, N);
    // z <- read (ping-pong the f16 buffers)
    { _Float16* a = zr16; zr16 = rr16; rr16 = a; a = zi16; zi16 = ri16; ri16 = a; }
  }

  // ---- VQ + output projection ----
  pack_concat<<<blocks((long long)ND2,256),256,0,stream>>>(accr, acci, zqin16, N, D);
  gemm(zqin16, cb16, scores, N, NS, D2, D2, D2, NS, 0,0,0, 1, nullptr, 1.f, 0.f);
  argmin_rows<<<N,256,0,stream>>>(scores, cbn, idx, NS);
  gather_cb<<<N,256,0,stream>>>(idx, codebook, zq16, D2);
  gemm(zq16, wo16, out, N, 256, D2, D2, D2, 256, 0,0,0, 1, bout, 1.f, 0.f);
}
